// Net_5488968204310
// MI455X (gfx1250) — matrix operand order repaired and checked by reading, not compiled, not run on hardware
//
#include <hip/hip_runtime.h>
#include <math.h>

// ---------------------------------------------------------------------------
// MI455X (gfx1250) fused attention + gate net.
// All GEMMs via v_wmma_f32_16x16x32_bf16 (wave32, fp32 accumulate).
// Flash-style online softmax: the [B,N,N] logits tensor is never materialized.
// A-tiles staged once per block in LDS (kills 8-24x redundant global reads);
// next key-block prefetched with global_prefetch during softmax/PV.
// ---------------------------------------------------------------------------

typedef __attribute__((ext_vector_type(16))) __bf16 v16bf;
typedef __attribute__((ext_vector_type(8)))  __bf16 v8bf;
typedef __attribute__((ext_vector_type(8)))  float  v8f;

#define D      128
#define NB     1024
#define BATCH  32
#define BN     (BATCH * NB)

static __device__ __forceinline__ v8f wmma_bf16(v16bf a, v16bf b, v8f c) {
  return __builtin_amdgcn_wmma_f32_16x16x32_bf16(false, a, false, b,
                                                 (short)0, c, false, false);
}

// A-fragment (16x32 bf16) for a row-major matrix with stride 128.
// Lane half hi=lane>>4; element e -> k = kt*32 + hi*16 + e (contiguous,
//                 matching the B-fragment loads).
static __device__ __forceinline__ v16bf afrag_from(const __bf16* rowbase, int kt, int hi) {
  const int k0 = kt * 32 + hi * 16;
  v8bf lo = *(const v8bf*)(rowbase + k0);
  v8bf hh = *(const v8bf*)(rowbase + k0 + 8);
  v16bf a;
#pragma unroll
  for (int i = 0; i < 8; ++i) { a[i] = lo[i]; a[i + 8] = hh[i]; }
  return a;
}

// Cooperative 16x128 bf16 tile stage (global row-major, stride 128 -> LDS).
// 256 threads x 8 contiguous bf16 each (16-byte ds/global ops).
static __device__ __forceinline__ void stage_tile(__bf16* dst, const __bf16* src, int tid) {
  const int r  = tid >> 4;
  const int c0 = (tid & 15) * 8;
  *(v8bf*)(dst + r * 128 + c0) = *(const v8bf*)(src + (size_t)r * D + c0);
}

// ---------------- conversion kernels ----------------
__global__ void k_cvt(const float* __restrict__ in, __bf16* __restrict__ out, int n) {
  int i = blockIdx.x * blockDim.x + threadIdx.x;
  if (i < n) out[i] = (__bf16)in[i];
}

// W row-major [k][n] -> Wt row-major [n][k] bf16 (so B-frags are contiguous)
__global__ void k_cvt_wT(const float* __restrict__ W, __bf16* __restrict__ Wt) {
  int i = blockIdx.x * blockDim.x + threadIdx.x;  // 0..16383
  int k = i >> 7, n = i & 127;
  Wt[n * D + k] = (__bf16)W[i];
}

// ---------------- fused QKV projection ----------------
// One block per 16-row tile; x A-frags staged in LDS once, reused for Q,K,V.
// Q pre-scaled by 1/sqrt(D); V stored transposed per batch [b][d][m].
__global__ void __launch_bounds__(256) k_qkv(
    const __bf16* __restrict__ xbf,
    const __bf16* __restrict__ WqT, const float* __restrict__ bq,
    const __bf16* __restrict__ WkT, const float* __restrict__ bk,
    const __bf16* __restrict__ WvT, const float* __restrict__ bv,
    __bf16* __restrict__ Q, __bf16* __restrict__ K, __bf16* __restrict__ Vt) {
  __shared__ __attribute__((aligned(32))) __bf16 Xsh[16 * 128];
  const int lane = threadIdx.x & 31;
  const int wv   = threadIdx.x >> 5;
  const int hi   = lane >> 4;
  const int l15  = lane & 15;
  const int row0 = blockIdx.x * 16;
  const int n    = wv * 16 + l15;

  stage_tile(Xsh, xbf + (size_t)row0 * D, threadIdx.x);
  __syncthreads();

  v16bf a[4];
#pragma unroll
  for (int kt = 0; kt < 4; ++kt) a[kt] = afrag_from(Xsh + l15 * 128, kt, hi);

  v8f aq = {}, ak = {}, av = {};
#pragma unroll
  for (int kt = 0; kt < 4; ++kt) {
    const int bo = n * D + kt * 32 + hi * 16;
    aq = wmma_bf16(a[kt], *(const v16bf*)(WqT + bo), aq);
    ak = wmma_bf16(a[kt], *(const v16bf*)(WkT + bo), ak);
    av = wmma_bf16(a[kt], *(const v16bf*)(WvT + bo), av);
  }
  const float bqv = bq[n], bkv = bk[n], bvv = bv[n];
  const float iscale = 0.08838834764831845f;  // 1/sqrt(128), folded into Q
#pragma unroll
  for (int j = 0; j < 8; ++j) {
    const int r    = row0 + j + hi * 8;
    const int bidx = r >> 10, mloc = r & 1023;
    Q[(size_t)r * D + n] = (__bf16)((aq[j] + bqv) * iscale);
    K[(size_t)r * D + n] = (__bf16)(ak[j] + bkv);
    Vt[(size_t)bidx * D * NB + (size_t)n * NB + mloc] = (__bf16)(av[j] + bvv);
  }
}

// ---------------- flash attention ----------------
// One block per (batch, 16-query tile). 8 waves: wave w owns key sub-tile w and
// output columns 16w..16w+15. Online softmax state (max/sum/alpha) in LDS.
__global__ void __launch_bounds__(256) k_attn(
    const __bf16* __restrict__ Q, const __bf16* __restrict__ K,
    const __bf16* __restrict__ Vt, __bf16* __restrict__ attn) {
  __shared__ __attribute__((aligned(32))) float  Ssh[16 * 128];
  __shared__ __attribute__((aligned(32))) __bf16 Psh[16 * 128];
  __shared__ __attribute__((aligned(32))) __bf16 Qsh[16 * 128];
  __shared__ float rowmax[16], rowsum[16], alphaS[16];

  const int lane = threadIdx.x & 31;
  const int wv   = threadIdx.x >> 5;
  const int hi   = lane >> 4;
  const int l15  = lane & 15;
  const int b    = blockIdx.x >> 6;
  const int row0 = (blockIdx.x & 63) * 16;

  const __bf16* Qb = Q  + (size_t)b * NB * D;
  const __bf16* Kb = K  + (size_t)b * NB * D;
  const __bf16* Vb = Vt + (size_t)b * D * NB;

  stage_tile(Qsh, Qb + (size_t)row0 * D, threadIdx.x);
  if (threadIdx.x < 16) { rowmax[threadIdx.x] = -3.4e38f; rowsum[threadIdx.x] = 0.f; }
  __syncthreads();

  v16bf qf[4];
#pragma unroll
  for (int kt = 0; kt < 4; ++kt) qf[kt] = afrag_from(Qsh + l15 * 128, kt, hi);

  v8f oacc = {};
  const int dcol = wv * 16 + l15;

  for (int kb = 0; kb < NB / 128; ++kb) {
    const int m0 = kb * 128 + wv * 16;
    // S = (Q/sqrtD) . K^T  for this wave's 16 keys
    v8f s = {};
#pragma unroll
    for (int kt = 0; kt < 4; ++kt) {
      v16bf bK = *(const v16bf*)(Kb + (size_t)(m0 + l15) * D + kt * 32 + hi * 16);
      s = wmma_bf16(qf[kt], bK, s);
    }
    // prefetch next key block (K rows and V columns) while we do softmax + PV
    if (kb + 1 < NB / 128) {
      __builtin_prefetch(Kb + (size_t)(m0 + 128 + l15) * D, 0, 0);
      __builtin_prefetch(Vb + (size_t)dcol * NB + (kb + 1) * 128, 0, 0);
    }
#pragma unroll
    for (int j = 0; j < 8; ++j)
      Ssh[(j + hi * 8) * 128 + wv * 16 + l15] = s[j];
    __syncthreads();

    {  // online softmax over this 128-key block: 16 lanes per row
      const int r  = threadIdx.x >> 4;
      const int c0 = threadIdx.x & 15;
      float mx = -3.4e38f;
      float vals[8];
#pragma unroll
      for (int i = 0; i < 8; ++i) { vals[i] = Ssh[r * 128 + c0 + i * 16]; mx = fmaxf(mx, vals[i]); }
#pragma unroll
      for (int off = 8; off >= 1; off >>= 1) mx = fmaxf(mx, __shfl_xor(mx, off, 16));
      const float om = rowmax[r];
      const float nm = fmaxf(om, mx);
      const float al = __expf(om - nm);
      float sum = 0.f;
#pragma unroll
      for (int i = 0; i < 8; ++i) {
        const float p = __expf(vals[i] - nm);
        sum += p;
        Psh[r * 128 + c0 + i * 16] = (__bf16)p;
      }
#pragma unroll
      for (int off = 8; off >= 1; off >>= 1) sum += __shfl_xor(sum, off, 16);
      if (c0 == 0) { rowsum[r] = rowsum[r] * al + sum; rowmax[r] = nm; alphaS[r] = al; }
    }
    __syncthreads();

    // rescale running output, then O += P . V
#pragma unroll
    for (int j = 0; j < 8; ++j) oacc[j] = oacc[j] * alphaS[j + hi * 8];
#pragma unroll
    for (int kt = 0; kt < 4; ++kt) {
      v16bf aP = afrag_from(Psh + l15 * 128, kt, hi);
      v16bf bV = *(const v16bf*)(Vb + (size_t)dcol * NB + kb * 128 + kt * 32 + hi * 16);
      oacc = wmma_bf16(aP, bV, oacc);
    }
    __syncthreads();
  }

#pragma unroll
  for (int j = 0; j < 8; ++j) {
    const int m = j + hi * 8;
    const float o = oacc[j] / rowsum[m];
    attn[(size_t)(b * NB + row0 + m) * D + dcol] = (__bf16)o;
  }
}

// ---------------- output projection + GateNet (5 fused GEMMs) ----------------
__global__ void __launch_bounds__(256) k_gate(
    const __bf16* __restrict__ xbf, const __bf16* __restrict__ attnbf,
    const float* __restrict__ xf,
    const __bf16* __restrict__ WoT,  const float* __restrict__ bo,
    const __bf16* __restrict__ Wo1T, const float* __restrict__ bo1,
    const __bf16* __restrict__ Wg1T, const float* __restrict__ bg1,
    const __bf16* __restrict__ Wg2T, const float* __restrict__ bg2,
    const __bf16* __restrict__ Wg3T, const float* __restrict__ bg3,
    float* __restrict__ out) {
  __shared__ __attribute__((aligned(32))) __bf16 Xsh[16 * 128];  // x tile
  __shared__ __attribute__((aligned(32))) __bf16 Ash[16 * 128];  // attn tile
  __shared__ __attribute__((aligned(32))) __bf16 Msh[16 * 128];  // msg tile
  __shared__ __attribute__((aligned(32))) __bf16 Gsh[16 * 128];  // gate_pre tile
  const int lane = threadIdx.x & 31;
  const int wv   = threadIdx.x >> 5;
  const int hi   = lane >> 4;
  const int l15  = lane & 15;
  const int row0 = blockIdx.x * 16;
  const int n    = wv * 16 + l15;

  stage_tile(Xsh, xbf    + (size_t)row0 * D, threadIdx.x);
  stage_tile(Ash, attnbf + (size_t)row0 * D, threadIdx.x);
  __syncthreads();

  v16bf ax[4], aa[4];
#pragma unroll
  for (int kt = 0; kt < 4; ++kt) {
    ax[kt] = afrag_from(Xsh + l15 * 128, kt, hi);
    aa[kt] = afrag_from(Ash + l15 * 128, kt, hi);
  }

  // msg = attn @ Wo + bo
  v8f cm = {};
#pragma unroll
  for (int kt = 0; kt < 4; ++kt)
    cm = wmma_bf16(aa[kt], *(const v16bf*)(WoT + n * D + kt * 32 + hi * 16), cm);
  float msg[8];
  const float bov = bo[n];
#pragma unroll
  for (int j = 0; j < 8; ++j) msg[j] = cm[j] + bov;

  // ret = x @ Wo1 + bo1 + msg
  v8f cr = {};
#pragma unroll
  for (int kt = 0; kt < 4; ++kt)
    cr = wmma_bf16(ax[kt], *(const v16bf*)(Wo1T + n * D + kt * 32 + hi * 16), cr);
  float ret[8];
  const float bo1v = bo1[n];
#pragma unroll
  for (int j = 0; j < 8; ++j) ret[j] = cr[j] + bo1v + msg[j];

  // re-fragment msg (C layout -> A layout) through LDS
#pragma unroll
  for (int j = 0; j < 8; ++j) Msh[(j + hi * 8) * 128 + n] = (__bf16)msg[j];
  __syncthreads();

  // gate_pre = relu(x @ Wg1 + bg1 + msg @ Wg2 + bg2)
  v8f cg = {};
#pragma unroll
  for (int kt = 0; kt < 4; ++kt) {
    cg = wmma_bf16(ax[kt], *(const v16bf*)(Wg1T + n * D + kt * 32 + hi * 16), cg);
    v16bf am = afrag_from(Msh + l15 * 128, kt, hi);
    cg = wmma_bf16(am, *(const v16bf*)(Wg2T + n * D + kt * 32 + hi * 16), cg);
  }
  const float bg12 = bg1[n] + bg2[n];
#pragma unroll
  for (int j = 0; j < 8; ++j)
    Gsh[(j + hi * 8) * 128 + n] = (__bf16)fmaxf(cg[j] + bg12, 0.f);
  __syncthreads();

  // gate = sigmoid(gate_pre @ Wg3 + bg3);  out = ret*gate + x*(1-gate)
  v8f cs = {};
#pragma unroll
  for (int kt = 0; kt < 4; ++kt) {
    v16bf ag = afrag_from(Gsh + l15 * 128, kt, hi);
    cs = wmma_bf16(ag, *(const v16bf*)(Wg3T + n * D + kt * 32 + hi * 16), cs);
  }
  const float bg3v = bg3[n];
#pragma unroll
  for (int j = 0; j < 8; ++j) {
    const int m = j + hi * 8;
    const float g  = 1.f / (1.f + __expf(-(cs[j] + bg3v)));
    const float xv = xf[(size_t)(row0 + m) * D + n];
    out[(size_t)(row0 + m) * D + n] = ret[j] * g + xv * (1.f - g);
  }
}

// ---------------------------------------------------------------------------
extern "C" void kernel_launch(void* const* d_in, const int* in_sizes, int n_in,
                              void* d_out, int out_size, void* d_ws, size_t ws_size,
                              hipStream_t stream) {
  const float* x   = (const float*)d_in[0];
  const float* Wq  = (const float*)d_in[1];  const float* bq  = (const float*)d_in[2];
  const float* Wk  = (const float*)d_in[3];  const float* bk  = (const float*)d_in[4];
  const float* Wv  = (const float*)d_in[5];  const float* bv  = (const float*)d_in[6];
  const float* Wo  = (const float*)d_in[7];  const float* bo  = (const float*)d_in[8];
  const float* Wo1 = (const float*)d_in[9];  const float* bo1 = (const float*)d_in[10];
  const float* Wg1 = (const float*)d_in[11]; const float* bg1 = (const float*)d_in[12];
  const float* Wg2 = (const float*)d_in[13]; const float* bg2 = (const float*)d_in[14];
  const float* Wg3 = (const float*)d_in[15]; const float* bg3 = (const float*)d_in[16];
  float* out = (float*)d_out;

  // workspace carve-out (~40.3 MB): bf16 x, Q, K, V^T, attn, 8 transposed weights
  char* ws = (char*)d_ws;
  size_t off = 0;
  auto alloc = [&](size_t bytes) -> char* {
    char* p = ws + off;
    off += (bytes + 255) & ~(size_t)255;
    return p;
  };
  __bf16* xbf    = (__bf16*)alloc((size_t)BN * D * 2);
  __bf16* Qbf    = (__bf16*)alloc((size_t)BN * D * 2);
  __bf16* Kbf    = (__bf16*)alloc((size_t)BN * D * 2);
  __bf16* Vtbf   = (__bf16*)alloc((size_t)BN * D * 2);
  __bf16* attnbf = (__bf16*)alloc((size_t)BN * D * 2);
  __bf16* WT[8];
  const float* Wsrc[8] = {Wq, Wk, Wv, Wo, Wo1, Wg1, Wg2, Wg3};
  for (int i = 0; i < 8; ++i) WT[i] = (__bf16*)alloc((size_t)D * D * 2);

  k_cvt<<<(BN * D + 255) / 256, 256, 0, stream>>>(x, xbf, BN * D);
  for (int i = 0; i < 8; ++i)
    k_cvt_wT<<<(D * D + 255) / 256, 256, 0, stream>>>(Wsrc[i], WT[i]);

  k_qkv<<<dim3(BN / 16), 256, 0, stream>>>(xbf, WT[0], bq, WT[1], bk, WT[2], bv,
                                           Qbf, Kbf, Vtbf);
  k_attn<<<dim3(BATCH * (NB / 16)), 256, 0, stream>>>(Qbf, Kbf, Vtbf, attnbf);
  k_gate<<<dim3(BN / 16), 256, 0, stream>>>(xbf, attnbf, x,
                                            WT[3], bo, WT[4], bo1, WT[5], bg1,
                                            WT[6], bg2, WT[7], bg3, out);
}
